// OT_Loss3_20177756357273
// MI455X (gfx1250) — matrix operand order repaired and checked by reading, not compiled, not run on hardware
//
#include <hip/hip_runtime.h>
#include <hip/hip_bf16.h>

typedef __attribute__((ext_vector_type(16))) _Float16 v16h;
typedef __attribute__((ext_vector_type(8)))  _Float16 v8h;
typedef __attribute__((ext_vector_type(8)))  float    v8f;

#define BATCH   8
#define NPTS    1024
#define DD      64
#define MM      4096
#define NITER   100
#define ACONST  (1.0f/1024.0f)
#define BLOCK   512
#define NWAVES  (BLOCK/32)

// ---- dynamic LDS layout (bytes) : total 301056 <= 320KB/WGP on CDNA5 ----
// Transposed storage: p-contiguous, so WMMA K-dim fragment reads are b128 loads.
#define KXT_OFF  0                          // f16 KxT[64][1024]  (128 KB)
#define KYT_OFF  (KXT_OFF + 64*1024*2)      // f16 KyT[64][1024]  (128 KB)
#define V_OFF    (KYT_OFF + 64*1024*2)      // f32 v[4096]        (16 KB)
#define U_OFF    (V_OFF  + 4096*4)          // f32 u[1024]        (4 KB)
#define U16_OFF  (U_OFF  + 1024*4)          // f16 u16[1024]      (2 KB)
#define VTB_OFF  (U16_OFF + 1024*2)         // f16 VtB[4096] packed B-frags (8 KB)
#define XN_OFF   (VTB_OFF + 4096*2)         // f32 [1024]
#define YN_OFF   (XN_OFF + 1024*4)          // f32 [1024]
#define SMEM_BYTES (YN_OFF + 1024*4)

__device__ __forceinline__ float block_sum(float x, float* red) {
  #pragma unroll
  for (int m = 16; m >= 1; m >>= 1) x += __shfl_xor(x, m, 32);
  const int tid = threadIdx.x;
  __syncthreads();
  if ((tid & 31) == 0) red[tid >> 5] = x;
  __syncthreads();
  float r = 0.f;
  if (tid == 0) {
    #pragma unroll
    for (int w = 0; w < NWAVES; ++w) r += red[w];
  }
  return r;   // valid on tid==0
}

__device__ __forceinline__ v8h lds_v8h(const _Float16* p) {
  return *(const v8h*)p;                    // 16B-aligned -> ds_load_b128
}

__global__ __launch_bounds__(BLOCK)
void sinkhorn_wmma(const float* __restrict__ g_nd, const float* __restrict__ g_ud,
                   const float* __restrict__ g_pts, const float* __restrict__ g_vp,
                   float* __restrict__ g_ws)
{
  extern __shared__ char smem[];
  _Float16* KxT = (_Float16*)(smem + KXT_OFF);   // KxT[mx*1024 + p]
  _Float16* KyT = (_Float16*)(smem + KYT_OFF);   // KyT[my*1024 + p]
  float*    vv  = (float*)   (smem + V_OFF);
  float*    uu  = (float*)   (smem + U_OFF);
  _Float16* u16 = (_Float16*)(smem + U16_OFF);
  _Float16* VtB = (_Float16*)(smem + VTB_OFF);   // packed phase-A B fragments
  float*    xs  = (float*)   (smem + XN_OFF);
  float*    ys  = (float*)   (smem + YN_OFF);
  __shared__ float red[NWAVES];
  __shared__ float fin[8];

  const int b    = blockIdx.x;
  const int tid  = threadIdx.x;
  const int lane = tid & 31;
  const int wave = tid >> 5;

  const float* nd  = g_nd + b*MM;
  const float* ud  = g_ud + b*MM;
  const float* vp  = g_vp + b*MM;
  const float* pts = g_pts + b*NPTS*2;

  // ---------------- setup: separable K = Ky (x) Kx, f16, p-contiguous ----------------
  for (int p = tid; p < NPTS; p += BLOCK) {
    float xn = pts[p*2 + 0]*(2.0f/512.0f) - 1.0f;
    float yn = pts[p*2 + 1]*(2.0f/512.0f) - 1.0f;
    xs[p] = xn; ys[p] = yn;
    #pragma unroll 4
    for (int j = 0; j < DD; ++j) {
      float c  = j*0.03125f - 0.984375f;          // cood[j]
      float dx = xn - c, dy = yn - c;
      KxT[j*NPTS + p] = (_Float16)__expf(-dx*dx*0.1f);
      KyT[j*NPTS + p] = (_Float16)__expf(-dy*dy*0.1f);
    }
  }
  for (int m = tid; m < MM; m += BLOCK) vv[m] = vp[m];   // v0 = v_pred
  __syncthreads();

  // ---------------- Sinkhorn: 100 iterations, WMMA-fed from b128 LDS loads ----------------
  for (int it = 0; it < NITER; ++it) {
    // Pack phase-A B operand directly in WMMA register order:
    // VtB[((kc*4+nt)*32 + l)*16 + e] = f16(v[my*64+mx]),
//   my = nt*16 + (l&15), mx = kc*32 + ((l&16)?8:0) + (e<8 ? e : e+8)
    for (int i = tid; i < MM; i += BLOCK) {
      int e   = i & 15;
      int l   = (i >> 4) & 31;
      int blk = i >> 9;
      int kc  = blk >> 2, nt = blk & 3;
      int my  = nt*16 + (l & 15);
      int mx  = kc*32 + ((l & 16) ? 8 : 0) + (e < 8 ? e : e + 8);
      VtB[i] = (_Float16)vv[my*DD + mx];
    }
    __syncthreads();

    // ---- Phase A: s[p] = sum_my Ky[p,my] * (Kx @ V^T)[p,my];  u = a/(s+eps)
    {
      const int n   = lane & 15;
      const int h   = (lane & 16) ? 8 : 0;    // A K-half
      const int mro = (lane & 16) ? 8 : 0;

      v16h Bf[2][4];
      #pragma unroll
      for (int kc = 0; kc < 2; ++kc)
        #pragma unroll
        for (int nt = 0; nt < 4; ++nt) {
          const _Float16* base = VtB + ((kc*4 + nt)*32 + lane)*16;
          v8h lo = lds_v8h(base), hi = lds_v8h(base + 8);
          #pragma unroll
          for (int e = 0; e < 8; ++e) { Bf[kc][nt][e] = lo[e]; Bf[kc][nt][8+e] = hi[e]; }
        }

      for (int t = 0; t < 64/NWAVES; ++t) {
        int mtile = wave*(64/NWAVES) + t;
        int pbase = mtile*16;
        int pA    = pbase + (lane & 15);
        v16h A0, A1;                          // strided columns of KxT (only 8 frags/wave/iter)
        #pragma unroll
        for (int e = 0; e < 8; ++e) {
          A0[e]   = KxT[(h + e)*NPTS + pA];
          A0[8+e] = KxT[(16 + h + e)*NPTS + pA];
          A1[e]   = KxT[(32 + h + e)*NPTS + pA];
          A1[8+e] = KxT[(48 + h + e)*NPTS + pA];
        }
        float partial[8];
        #pragma unroll
        for (int r = 0; r < 8; ++r) partial[r] = 0.f;
        #pragma unroll
        for (int nt = 0; nt < 4; ++nt) {
          v8f acc = {0.f,0.f,0.f,0.f,0.f,0.f,0.f,0.f};
          acc = __builtin_amdgcn_wmma_f32_16x16x32_f16(false, A0, false, Bf[0][nt], (short)0, acc, false, false);
          acc = __builtin_amdgcn_wmma_f32_16x16x32_f16(false, A1, false, Bf[1][nt], (short)0, acc, false, false);
          // Ky weights: 8 contiguous f16 in p -> one b128 load
          v8h kyv = lds_v8h(KyT + (nt*16 + n)*NPTS + pbase + mro);
          #pragma unroll
          for (int r = 0; r < 8; ++r) partial[r] += acc[r] * (float)kyv[r];
        }
        #pragma unroll
        for (int r = 0; r < 8; ++r) {         // reduce over N within 16-lane halves
          float s = partial[r];
          s += __shfl_xor(s, 8, 16);
          s += __shfl_xor(s, 4, 16);
          s += __shfl_xor(s, 2, 16);
          s += __shfl_xor(s, 1, 16);
          partial[r] = s;
        }
        if ((lane & 15) == 0) {
          #pragma unroll
          for (int r = 0; r < 8; ++r) {
            float uval = ACONST / (partial[r] + 1e-16f);
            uu [pbase + mro + r] = uval;
            u16[pbase + mro + r] = (_Float16)uval;
          }
        }
      }
    }
    __syncthreads();

    // ---- Phase B: T[my,mx] = sum_p Kx[p,mx]*Ky[p,my]*u[p];  v = b/(T+eps)
    {
      const int n   = lane & 15;
      const int kh  = (lane & 16) ? 8 : 0;
      const int h   = (lane & 16) ? 8  : 0;
      const int m   = lane & 15;
      const int mro = (lane & 16) ? 8 : 0;
      for (int tt = 0; tt < 16/NWAVES; ++tt) {
        int tile = wave*(16/NWAVES) + tt;
        int mxt = tile >> 2, myt = tile & 3;
        v8f acc = {0.f,0.f,0.f,0.f,0.f,0.f,0.f,0.f};
        for (int ch = 0; ch < 32; ++ch) {
          int pb = ch*32;
          // A: rows of KxT (mx-major), K contiguous in p -> 2x b128
          const _Float16* rA = KxT + (mxt*16 + m)*NPTS + pb;
          v8h a0 = lds_v8h(rA + h), a1 = lds_v8h(rA + 16 + h);
          v16h Af;
          #pragma unroll
          for (int e = 0; e < 8; ++e) { Af[e] = a0[e]; Af[8+e] = a1[e]; }
          // B: Ky row (contiguous in p) * u16 (contiguous, broadcast) -> pk_mul_f16
          const _Float16* rB = KyT + (myt*16 + n)*NPTS + pb + kh;
          v8h k0 = lds_v8h(rB),            k1 = lds_v8h(rB + 16);
          v8h w0 = lds_v8h(u16 + pb + kh), w1 = lds_v8h(u16 + pb + kh + 16);
          v8h p0 = k0 * w0, p1 = k1 * w1;
          v16h Bfv;
          #pragma unroll
          for (int e = 0; e < 8; ++e) { Bfv[e] = p0[e]; Bfv[8+e] = p1[e]; }
          acc = __builtin_amdgcn_wmma_f32_16x16x32_f16(false, Af, false, Bfv, (short)0, acc, false, false);
        }
        #pragma unroll
        for (int r = 0; r < 8; ++r) {
          int mx = mxt*16 + mro + r;
          int my = myt*16 + n;
          int mi = my*DD + mx;
          vv[mi] = nd[mi] / (acc[r] + 1e-16f);
        }
      }
    }
    __syncthreads();
  }

  // ---------------- final reductions ----------------
  float ot=0.f, sc=0.f, sb=0.f, dvv=0.f, nv2=0.f, nvp2=0.f;
  for (int m = tid; m < MM; m += BLOCK) {
    float vm   = vv[m];
    float beta = 10.f*__logf(vm + 1e-16f);
    float ndv = nd[m], udv = ud[m], vpm = vp[m];
    ot += ndv*beta; sc += udv; sb += udv*beta;
    dvv += vpm*vm; nv2 += vm*vm; nvp2 += vpm*vpm;
  }
  { float r = block_sum(ot,  red); if (tid==0) fin[0]=r; }
  { float r = block_sum(sc,  red); if (tid==0) fin[1]=r; }
  { float r = block_sum(sb,  red); if (tid==0) fin[2]=r; }
  { float r = block_sum(dvv, red); if (tid==0) fin[3]=r; }
  { float r = block_sum(nv2, red); if (tid==0) fin[4]=r; }
  { float r = block_sum(nvp2,red); if (tid==0) fin[5]=r; }
  __syncthreads();
  const float scv = fin[1], sbv = fin[2];
  const float c2 = 1.f/(scv*scv + 1e-8f);
  float lg = 0.f;
  for (int m = tid; m < MM; m += BLOCK) {
    float beta = 10.f*__logf(vv[m] + 1e-16f);
    lg += ud[m]*(scv*c2*beta - sbv*c2);       // sum(ud * im_grad)
  }
  float lgs = block_sum(lg, red);

  // dense one-time pass: u_pred = a/(K@vp+eps) and wd = sum(dis*P)
  float duu=0.f, nup=0.f, nu2=0.f, wda=0.f;
  for (int p = tid; p < NPTS; p += BLOCK) {
    float kv=0.f, wdp=0.f;
    const float xnp = xs[p], ynp = ys[p];
    for (int my = 0; my < DD; ++my) {
      float ky = (float)KyT[my*NPTS + p];
      float cy = my*0.03125f - 0.984375f;
      float dy = ynp - cy; float yd = dy*dy;
      const float* vrow  = vv + my*DD;
      const float* vprow = vp + my*DD;
      __builtin_prefetch(vprow, 0, 0);          // global_prefetch_b8
      #pragma unroll 4
      for (int mx = 0; mx < DD; ++mx) {
        float K = ky * (float)KxT[mx*NPTS + p];
        kv += K * vprow[mx];
        float cx = mx*0.03125f - 0.984375f;
        float dx = xnp - cx;
        wdp += (yd + dx*dx) * K * vrow[mx];
      }
    }
    float upd = ACONST/(kv + 1e-16f);
    float up  = uu[p];
    duu += upd*up; nup += upd*upd; nu2 += up*up; wda += up*wdp;
  }
  float sduu = block_sum(duu, red);
  float snup = block_sum(nup, red);
  float snu2 = block_sum(nu2, red);
  float swda = block_sum(wda, red);

  if (tid == 0) {
    float cos1 = fin[3]/(fmaxf(sqrtf(fin[5]),1e-8f)*fmaxf(sqrtf(fin[4]),1e-8f));
    float cos2 = sduu /(fmaxf(sqrtf(snup),1e-8f)*fmaxf(sqrtf(snu2),1e-8f));
    float lossb = (1.f - cos1) + (1.f - cos2);
    g_ws[b*3+0] = lgs + lossb;   // loss_i
    g_ws[b*3+1] = swda;          // wd_i
    g_ws[b*3+2] = fin[0];        // ot_i
  }
}

__global__ void finalize_sum(const float* __restrict__ ws, float* __restrict__ out) {
  int k = threadIdx.x;
  if (k < 3) {
    float s = 0.f;
    for (int b = 0; b < BATCH; ++b) s += ws[b*3 + k];
    out[k] = s;
  }
}

extern "C" void kernel_launch(void* const* d_in, const int* in_sizes, int n_in,
                              void* d_out, int out_size, void* d_ws, size_t ws_size,
                              hipStream_t stream) {
  const float* nd  = (const float*)d_in[0];   // normed_density  [8,1,64,64]
  const float* ud  = (const float*)d_in[1];   // unnormed_density
  const float* pts = (const float*)d_in[2];   // points [8,1024,2]
  const float* vp  = (const float*)d_in[3];   // v_pred [8,4096]
  float* ws = (float*)d_ws;

  (void)in_sizes; (void)n_in; (void)out_size; (void)ws_size;
  (void)hipFuncSetAttribute((const void*)sinkhorn_wmma,
                            hipFuncAttributeMaxDynamicSharedMemorySize, SMEM_BYTES);
  sinkhorn_wmma<<<dim3(BATCH), dim3(BLOCK), SMEM_BYTES, stream>>>(nd, ud, pts, vp, ws);
  finalize_sum<<<dim3(1), dim3(32), 0, stream>>>(ws, (float*)d_out);
}
